// GraphModel_87737591922707
// MI455X (gfx1250) — matrix operand order repaired and checked by reading, not compiled, not run on hardware
//
#include <hip/hip_runtime.h>
#include <hip/hip_bf16.h>
#include <math.h>

// ---------------------------------------------------------------------------
// Problem constants (from the reference)
// ---------------------------------------------------------------------------
#define TT     12
#define NN     512
#define F_IN   32
#define F_OUT  8
#define EE     16384
#define HH     4096          // N * F_OUT
#define H4     16384         // 4*H

typedef __attribute__((ext_vector_type(16))) __bf16 v16bf;
typedef __attribute__((ext_vector_type(8)))  __bf16 v8bf;
typedef __attribute__((ext_vector_type(4)))  __bf16 v4bf;
typedef __attribute__((ext_vector_type(8)))  float  v8f;

static __device__ __forceinline__ __bf16 f32_to_bf16(float f) {
    unsigned u = __builtin_bit_cast(unsigned, f);
    u += 0x7FFFu + ((u >> 16) & 1u);              // round-to-nearest-even
    unsigned short h = (unsigned short)(u >> 16);
    return __builtin_bit_cast(__bf16, h);
}

static __device__ __forceinline__ v16bf load_a32(const __bf16* p) {
    v8bf lo = *(const v8bf*)(p);
    v8bf hi = *(const v8bf*)(p + 16);
    return __builtin_shufflevector(lo, hi, 0,1,2,3,4,5,6,7,8,9,10,11,12,13,14,15);
}

// ---------------------------------------------------------------------------
// f32 -> bf16 weight conversion (vectorized, grid covers n/4 threads)
// ---------------------------------------------------------------------------
__global__ void k_cvt_bf16(const float* __restrict__ src, __bf16* __restrict__ dst, int n) {
    int i = (blockIdx.x * blockDim.x + threadIdx.x) * 4;
    if (i >= n) return;
    float4 v = *(const float4*)(src + i);
    v4bf o;
    o[0] = f32_to_bf16(v.x); o[1] = f32_to_bf16(v.y);
    o[2] = f32_to_bf16(v.z); o[3] = f32_to_bf16(v.w);
    *(v4bf*)(dst + i) = o;
}

// ---------------------------------------------------------------------------
// GCN phase (tiny: ~2MB total traffic)
// ---------------------------------------------------------------------------
__global__ void k_gcn_xw(const float* __restrict__ x, const float* __restrict__ Wg,
                         float* __restrict__ hfeat) {
    int idx = blockIdx.x * blockDim.x + threadIdx.x;   // [T*N*F_OUT)
    if (idx >= TT * NN * F_OUT) return;
    int fo = idx & (F_OUT - 1);
    int tn = idx >> 3;
    const float* xr = x + (size_t)tn * F_IN;
    float s = 0.f;
#pragma unroll
    for (int k = 0; k < F_IN; ++k) s += xr[k] * Wg[k * F_OUT + fo];
    hfeat[idx] = s;
}

__global__ void k_deg_init(float* __restrict__ deg) {
    int i = blockIdx.x * blockDim.x + threadIdx.x;
    if (i < TT * NN) deg[i] = 1.0f;                    // self-loop weight
}

__global__ void k_deg_scatter(const int* __restrict__ ei, const float* __restrict__ ew,
                              float* __restrict__ deg) {
    int idx = blockIdx.x * blockDim.x + threadIdx.x;   // [T*E)
    if (idx >= TT * EE) return;
    int t = idx / EE, e = idx - t * EE;
    int col = ei[(size_t)t * 2 * EE + EE + e];
    atomicAdd(&deg[t * NN + col], ew[(size_t)t * EE + e]);
}

__global__ void k_dinv(float* __restrict__ deg) {
    int i = blockIdx.x * blockDim.x + threadIdx.x;
    if (i >= TT * NN) return;
    float d = deg[i];
    deg[i] = (d > 0.f) ? rsqrtf(d) : 0.f;
}

__global__ void k_feat_self(const float* __restrict__ dinv, const float* __restrict__ hfeat,
                            float* __restrict__ feats) {
    int idx = blockIdx.x * blockDim.x + threadIdx.x;   // [T*N*F_OUT)
    if (idx >= TT * NN * F_OUT) return;
    int tn = idx >> 3;
    float di = dinv[tn];
    feats[idx] = di * di * hfeat[idx];
}

__global__ void k_edge_scatter(const int* __restrict__ ei, const float* __restrict__ ew,
                               const float* __restrict__ dinv, const float* __restrict__ hfeat,
                               float* __restrict__ feats) {
    int idx = blockIdx.x * blockDim.x + threadIdx.x;   // [T*E)
    if (idx >= TT * EE) return;
    int t = idx / EE, e = idx - t * EE;
    int row = ei[(size_t)t * 2 * EE + e];
    int col = ei[(size_t)t * 2 * EE + EE + e];
    float norm = dinv[t * NN + row] * ew[(size_t)t * EE + e] * dinv[t * NN + col];
    const float* hsrc = hfeat + ((size_t)t * NN + row) * F_OUT;
    float* fdst = feats + (size_t)t * HH + col * F_OUT;
#pragma unroll
    for (int f = 0; f < F_OUT; ++f) atomicAdd(&fdst[f], norm * hsrc[f]);
}

__global__ void k_bias_relu(const float* __restrict__ bg, float* __restrict__ feats,
                            __bf16* __restrict__ feats_bf) {
    int idx = blockIdx.x * blockDim.x + threadIdx.x;   // [T*H)
    if (idx >= TT * HH) return;
    float v = fmaxf(feats[idx] + bg[idx & (F_OUT - 1)], 0.f);
    feats[idx] = v;
    feats_bf[idx] = f32_to_bf16(v);
}

__global__ void k_init_state(float* __restrict__ h, float* __restrict__ c,
                             __bf16* __restrict__ h_bf) {
    int i = blockIdx.x * blockDim.x + threadIdx.x;
    if (i >= HH) return;
    h[i] = 0.f; c[i] = 0.f;
    h_bf[i] = __builtin_bit_cast(__bf16, (unsigned short)0);
}

// ---------------------------------------------------------------------------
// Batched input-gate GEMM:  G_in[t, j] = b_ih[j] + b_hh[j] + feats[t,:] . W_ih[j,:]
// A = feats (16x32 bf16 tile, rows 12..15 zero — zeroed ONCE before the loop,
// EXEC-masked loads overwrite only live lanes), B = W_ih^T tile, C = f32.
// One wave per 16-wide j-tile; K-loop unrolled x2 with dual accumulators.
// ---------------------------------------------------------------------------
__global__ void k_gemm_in(const __bf16* __restrict__ feats_bf,   // [T, H]
                          const __bf16* __restrict__ Wih_bf,     // [4H, H]
                          const float* __restrict__ b_ih, const float* __restrict__ b_hh,
                          float* __restrict__ G_in) {             // [T, 4H]
    const int lane = threadIdx.x & 31;
    const int wave = threadIdx.x >> 5;
    const int tile = blockIdx.x * (blockDim.x >> 5) + wave;       // [0, 1024)
    const int j0 = tile * 16;
    const int j = j0 + (lane & 15);
    const int m = lane & 15;                                       // A row (timestep)
    const int koffA = (lane < 16) ? 0 : 8;
    const int koffB = (lane < 16) ? 0 : 8;
    const bool a_live = (m < TT);

    const __bf16* wrow = Wih_bf + (size_t)j * HH + koffB;
    const __bf16* arow = feats_bf + (size_t)m * HH + koffA;

    v16bf a0 = {}, a1 = {};            // dead lanes stay zero for the whole loop
    v8f acc0 = {}, acc1 = {};
    for (int k0 = 0; k0 < HH; k0 += 64) {
        if (a_live) {
            a0 = load_a32(arow + k0);
            a1 = load_a32(arow + k0 + 32);
        }
        v16bf b0 = load_a32(wrow + k0);
        v16bf b1 = load_a32(wrow + k0 + 32);
        __builtin_prefetch(wrow + k0 + 1024, 0, 0);                // stream ahead
        acc0 = __builtin_amdgcn_wmma_f32_16x16x32_bf16(
                   false, a0, false, b0, (short)0, acc0, false, false);
        acc1 = __builtin_amdgcn_wmma_f32_16x16x32_bf16(
                   false, a1, false, b1, (short)0, acc1, false, false);
    }
    const float bias = b_ih[j] + b_hh[j];
#pragma unroll
    for (int r = 0; r < 8; ++r) {
        int row = (lane < 16) ? r : (8 + r);
        if (row < TT) G_in[(size_t)row * H4 + j] = acc0[r] + acc1[r] + bias;
    }
}

// ---------------------------------------------------------------------------
// Recurrent GEMV (WMMA, M=1 padded to 16):
//   gates[j] = G_in[t, j] + h . W_hh[j, :]
// One wave per 16-wide j-tile; B streams W_hh (bf16, L2-resident after step 0).
// A rows 1..15 zeroed once; only lanes 0/16 refresh row 0 each k-step.
// ---------------------------------------------------------------------------
__global__ void k_gemv_hh(const __bf16* __restrict__ h_bf,        // [H]
                          const __bf16* __restrict__ Whh_bf,      // [4H, H]
                          const float* __restrict__ G_in_t,       // [4H]
                          float* __restrict__ gates) {            // [4H]
    const int lane = threadIdx.x & 31;
    const int wave = threadIdx.x >> 5;
    const int tile = blockIdx.x * (blockDim.x >> 5) + wave;       // [0, 1024)
    const int j0 = tile * 16;
    const int j = j0 + (lane & 15);
    const int koffA = (lane < 16) ? 0 : 8;
    const int koffB = (lane < 16) ? 0 : 8;
    const bool a_live = (lane & 15) == 0;                          // only row M=0 valid

    const __bf16* wrow = Whh_bf + (size_t)j * HH + koffB;
    const __bf16* hrow = h_bf + koffA;

    v16bf a0 = {}, a1 = {};            // rows 1..15 stay zero for the whole loop
    v8f acc0 = {}, acc1 = {};
    for (int k0 = 0; k0 < HH; k0 += 64) {
        if (a_live) {
            a0 = load_a32(hrow + k0);
            a1 = load_a32(hrow + k0 + 32);
        }
        v16bf b0 = load_a32(wrow + k0);
        v16bf b1 = load_a32(wrow + k0 + 32);
        __builtin_prefetch(wrow + k0 + 1024, 0, 0);
        acc0 = __builtin_amdgcn_wmma_f32_16x16x32_bf16(
                   false, a0, false, b0, (short)0, acc0, false, false);
        acc1 = __builtin_amdgcn_wmma_f32_16x16x32_bf16(
                   false, a1, false, b1, (short)0, acc1, false, false);
    }
    if (lane < 16) gates[j] = acc0[0] + acc1[0] + G_in_t[j];
}

// ---------------------------------------------------------------------------
// Pointwise LSTM cell update (torch gate order i,f,g,o)
// ---------------------------------------------------------------------------
__global__ void k_lstm_point(const float* __restrict__ gates, float* __restrict__ h,
                             float* __restrict__ c, __bf16* __restrict__ h_bf,
                             float* __restrict__ out_c, float* __restrict__ out_h,
                             int write_out) {
    int i = blockIdx.x * blockDim.x + threadIdx.x;
    if (i >= HH) return;
    float gi = gates[i];
    float gf = gates[HH + i];
    float gg = gates[2 * HH + i];
    float go = gates[3 * HH + i];
    float si = 1.f / (1.f + expf(-gi));
    float sf = 1.f / (1.f + expf(-gf));
    float so = 1.f / (1.f + expf(-go));
    float cc = sf * c[i] + si * tanhf(gg);
    float hh = so * tanhf(cc);
    c[i] = cc;
    h[i] = hh;
    h_bf[i] = f32_to_bf16(hh);
    if (write_out) { out_c[i] = cc; out_h[i] = hh; }
}

// ---------------------------------------------------------------------------
// Launch: GCN -> bf16 conversion -> batched input GEMM -> 12 x (GEMV + cell)
// ---------------------------------------------------------------------------
extern "C" void kernel_launch(void* const* d_in, const int* in_sizes, int n_in,
                              void* d_out, int out_size, void* d_ws, size_t ws_size,
                              hipStream_t stream) {
    (void)in_sizes; (void)n_in; (void)out_size; (void)ws_size;

    const float* x     = (const float*)d_in[0];
    const int*   ei    = (const int*)  d_in[1];
    const float* ew    = (const float*)d_in[2];
    const float* W_gcn = (const float*)d_in[3];
    const float* b_gcn = (const float*)d_in[4];
    const float* W_ih  = (const float*)d_in[5];
    const float* W_hh  = (const float*)d_in[6];
    const float* b_ih  = (const float*)d_in[7];
    const float* b_hh  = (const float*)d_in[8];

    float* out   = (float*)d_out;
    float* out_c = out;            // reference returns (c, h)
    float* out_h = out + HH;

    // Workspace layout (bytes)
    const size_t WBYTES = (size_t)H4 * HH * 2;                 // 134,217,728
    char* ws = (char*)d_ws;
    __bf16* Wih_bf   = (__bf16*)(ws);
    __bf16* Whh_bf   = (__bf16*)(ws + WBYTES);
    float*  hfeat    = (float*) (ws + 2 * WBYTES);                             // T*N*F_OUT
    float*  dinv     = (float*) (ws + 2 * WBYTES + 196608);                    // T*N (deg then dinv)
    float*  feats    = (float*) (ws + 2 * WBYTES + 196608 + 24576);            // T*H f32
    __bf16* feats_bf = (__bf16*)(ws + 2 * WBYTES + 196608 + 24576 + 196608);   // T*H bf16
    float*  G_in     = (float*) (ws + 2 * WBYTES + 196608 + 24576 + 196608 + 98304);   // T*4H
    float*  gates    = (float*) (ws + 2 * WBYTES + 196608 + 24576 + 196608 + 98304 + 786432);
    float*  hbuf     = (float*) (ws + 2 * WBYTES + 196608 + 24576 + 196608 + 98304 + 786432 + 65536);
    float*  cbuf     = (float*) ((char*)hbuf + HH * 4);
    __bf16* h_bf     = (__bf16*)((char*)cbuf + HH * 4);

    const int B = 256;

    // Weight conversion (12x-streamed W_hh becomes 134MB -> L2 resident)
    {
        const int n = H4 * HH;                     // 67,108,864 per matrix
        const int blocks = n / (B * 4);            // 65536
        k_cvt_bf16<<<blocks, B, 0, stream>>>(W_ih, Wih_bf, n);
        k_cvt_bf16<<<blocks, B, 0, stream>>>(W_hh, Whh_bf, n);
    }

    // GCN phase
    k_gcn_xw     <<<(TT * NN * F_OUT + B - 1) / B, B, 0, stream>>>(x, W_gcn, hfeat);
    k_deg_init   <<<(TT * NN + B - 1) / B,        B, 0, stream>>>(dinv);
    k_deg_scatter<<<(TT * EE + B - 1) / B,        B, 0, stream>>>(ei, ew, dinv);
    k_dinv       <<<(TT * NN + B - 1) / B,        B, 0, stream>>>(dinv);
    k_feat_self  <<<(TT * NN * F_OUT + B - 1) / B, B, 0, stream>>>(dinv, hfeat, feats);
    k_edge_scatter<<<(TT * EE + B - 1) / B,       B, 0, stream>>>(ei, ew, dinv, hfeat, feats);
    k_bias_relu  <<<(TT * HH + B - 1) / B,        B, 0, stream>>>(b_gcn, feats, feats_bf);
    k_init_state <<<(HH + B - 1) / B,             B, 0, stream>>>(hbuf, cbuf, h_bf);

    // Batched input-side gates: 1024 j-tiles, 8 waves/block
    k_gemm_in<<<128, B, 0, stream>>>(feats_bf, Wih_bf, b_ih, b_hh, G_in);

    // Sequential recurrence
    for (int t = 0; t < TT; ++t) {
        k_gemv_hh<<<128, B, 0, stream>>>(h_bf, Whh_bf, G_in + (size_t)t * H4, gates);
        k_lstm_point<<<(HH + B - 1) / B, B, 0, stream>>>(gates, hbuf, cbuf, h_bf,
                                                         out_c, out_h, t == TT - 1 ? 1 : 0);
    }
}
